// SAMG_50268297232812
// MI455X (gfx1250) — compile-verified, matrix operand order checked by reading, not run on hardware
//
#include <hip/hip_runtime.h>
#include <hip/hip_bf16.h>

typedef __attribute__((ext_vector_type(16))) __bf16 v16bf;
typedef __attribute__((ext_vector_type(8)))  float  v8f;

namespace {
constexpr int NTOK = 4096;     // B*L
constexpr int DM   = 1024;     // d_model
constexpr int DK   = 128;      // d_key
constexpr int MN   = 32768;    // graph nodes
constexpr int TK   = 32;       // top-k
constexpr int EM   = 8;        // edges per node
constexpr float SCALE = 11.313708498984761f;  // sqrt(d_key)
}

// ---------------------------------------------------------------------------
// bf16 WMMA fragment loader (global or LDS source): row-major [rows, ld],
// 16x32 (rows x K) tile. ISA layout (CDNA5 ISA): lane L holds row (L&15);
// K halves split by lane>>4; pair p covers K = (p&3)*2 + (p>>2)*16 +
// (lane>>4)*8. Same layout for A (MxK) and B when stored row-major [N, K].
// ---------------------------------------------------------------------------
__device__ __forceinline__ v16bf load_frag_bf16(const __bf16* base, int row0,
                                                int k0, int ld, int lane) {
  const __bf16* p = base + (size_t)(row0 + (lane & 15)) * ld
                         + (size_t)(k0 + ((lane >> 4) << 3));
  v16bf f;
#pragma unroll
  for (int e = 0; e < 8; ++e) {
    const int kb = ((e & 3) << 1) | ((e >> 2) << 4);
    f[2 * e]     = p[kb];
    f[2 * e + 1] = p[kb + 1];
  }
  return f;
}

// ---------------------------------------------------------------------------
// C[M,N] = A[M,K] * B[N,K]^T (bf16 in, fp32 out), double-buffered LDS B panel.
// Block = 256 thr (8 waves) computes a 128x64 macro-tile; per k-step (K+=32)
// a 4 KB B panel is staged once and shared by all 8 waves. Software pipeline:
// fetch panel s+1 (global, LOADcnt) -> consume panel s (ds frags + 4 WMMAs)
// -> store panel s+1 to the other buffer -> ONE barrier. Next A fragment is
// also prefetched into registers during compute.
// ---------------------------------------------------------------------------
__global__ __launch_bounds__(256) void wmma_gemm_bf16(
    const __bf16* __restrict__ A, int lda,
    const __bf16* __restrict__ Bm, int ldb,
    float* __restrict__ C, int ldc,
    int mrows, int ncols, int kdim) {
  __shared__ __bf16 bt[2][64 * 32];  // 2 x 4 KB B panels
  const int lane = threadIdx.x & 31;
  const int wave = threadIdx.x >> 5;
  const int nbn  = ncols >> 6;
  const int bm   = blockIdx.x / nbn;
  const int bn   = blockIdx.x - bm * nbn;
  const int m0   = bm * 128 + wave * 16;
  const int n0   = bn * 64;
  const int trow = threadIdx.x >> 2;   // 0..63 : N row in panel
  const int tch  = threadIdx.x & 3;    // 0..3  : 16B chunk in row
  if (m0 >= mrows) return;

  const __bf16* srcrow = Bm + (size_t)(n0 + trow) * ldb;
  const int nsteps = kdim >> 5;

  // prologue: stage panel 0
  uint4 stage = ((const uint4*)srcrow)[tch];
  ((uint4*)bt[0])[threadIdx.x] = stage;
  __syncthreads();
  v16bf a = load_frag_bf16(A, m0, 0, lda, lane);

  v8f acc[4] = {{}, {}, {}, {}};
  for (int s = 0; s < nsteps; ++s) {
    const int  cur  = s & 1;
    const bool more = (s + 1) < nsteps;
    if (more) {                                   // fetch next panel (in flight)
      stage = ((const uint4*)(srcrow + (size_t)(s + 1) * 32))[tch];
      if (s + 2 < nsteps)
        __builtin_prefetch((const char*)(srcrow + (size_t)(s + 2) * 32) + tch * 16, 0, 1);
    }
    // consume current panel: preload all 4 B fragments, then 4 WMMAs
    const v16bf b0 = load_frag_bf16(bt[cur],  0, 0, 32, lane);
    const v16bf b1 = load_frag_bf16(bt[cur], 16, 0, 32, lane);
    const v16bf b2 = load_frag_bf16(bt[cur], 32, 0, 32, lane);
    const v16bf b3 = load_frag_bf16(bt[cur], 48, 0, 32, lane);
    const v16bf a_cur = a;
    if (more) {
      ((uint4*)bt[cur ^ 1])[threadIdx.x] = stage; // park next panel
      a = load_frag_bf16(A, m0, (s + 1) * 32, lda, lane);
    }
    acc[0] = __builtin_amdgcn_wmma_f32_16x16x32_bf16(false, a_cur, false, b0,
                                                     (short)0, acc[0], false, false);
    acc[1] = __builtin_amdgcn_wmma_f32_16x16x32_bf16(false, a_cur, false, b1,
                                                     (short)0, acc[1], false, false);
    acc[2] = __builtin_amdgcn_wmma_f32_16x16x32_bf16(false, a_cur, false, b2,
                                                     (short)0, acc[2], false, false);
    acc[3] = __builtin_amdgcn_wmma_f32_16x16x32_bf16(false, a_cur, false, b3,
                                                     (short)0, acc[3], false, false);
    __syncthreads();                               // one barrier per k-step
  }
  const int colb = lane & 15;
  const int r0   = m0 + ((lane >> 4) << 3);
#pragma unroll
  for (int j = 0; j < 4; ++j)
#pragma unroll
    for (int v = 0; v < 8; ++v)
      C[(size_t)(r0 + v) * ldc + n0 + j * 16 + colb] = acc[j][v];
}

// ---------------------------------------------------------------------------
// fp32 -> bf16 grid-stride convert
// ---------------------------------------------------------------------------
__global__ __launch_bounds__(256) void f2bf_kernel(const float* __restrict__ s,
                                                   __bf16* __restrict__ d, int n) {
  for (int i = blockIdx.x * blockDim.x + threadIdx.x; i < n;
       i += gridDim.x * blockDim.x)
    d[i] = (__bf16)s[i];
}

// h[4096,1024] -> first half of hr_bf[4096,2048]
__global__ __launch_bounds__(256) void h_to_hr_kernel(const float* __restrict__ h,
                                                      __bf16* __restrict__ hr) {
  for (int i = blockIdx.x * blockDim.x + threadIdx.x; i < NTOK * DM;
       i += gridDim.x * blockDim.x) {
    const int row = i >> 10, col = i & (DM - 1);
    hr[(size_t)row * (2 * DM) + col] = (__bf16)h[i];
  }
}

// ---------------------------------------------------------------------------
// Row-normalize node_keys -> bf16 K-hat. One block (128 thr) per node.
// ---------------------------------------------------------------------------
__global__ __launch_bounds__(128) void norm_keys_kernel(const float* __restrict__ nk,
                                                        __bf16* __restrict__ kbf) {
  const int n = blockIdx.x, d = threadIdx.x;
  __shared__ float red[128];
  const float x = nk[(size_t)n * DK + d];
  red[d] = x * x;
  __syncthreads();
  for (int s = 64; s > 0; s >>= 1) {
    if (d < s) red[d] += red[d + s];
    __syncthreads();
  }
  const float inv = rsqrtf(red[0] + 1e-12f);
  kbf[(size_t)n * DK + d] = (__bf16)(x * inv);
}

// Row-normalize q -> fp32 q-hat (hop dots) + bf16 q-hat (WMMA scores)
__global__ __launch_bounds__(128) void norm_q_kernel(const float* __restrict__ q,
                                                     float* __restrict__ qn,
                                                     __bf16* __restrict__ qbf) {
  const int t = blockIdx.x, d = threadIdx.x;
  __shared__ float red[128];
  const float x = q[(size_t)t * DK + d];
  red[d] = x * x;
  __syncthreads();
  for (int s = 64; s > 0; s >>= 1) {
    if (d < s) red[d] += red[d + s];
    __syncthreads();
  }
  const float inv = rsqrtf(red[0] + 1e-12f);
  const float v = x * inv;
  qn[(size_t)t * DK + d]  = v;
  qbf[(size_t)t * DK + d] = (__bf16)v;
}

// ---------------------------------------------------------------------------
// Fused scores + top-32. Block = 128 tokens (8 waves x 16 tokens; each wave's
// 4 A-fragments live in registers). All waves sweep the 2048 key tiles in
// lockstep; each 16x128 bf16 key tile (4 KB) is staged in a double-buffered
// LDS slot and consumed by all 8 waves (8x less L2 traffic). Pipeline: fetch
// tile nt+1 -> 4 WMMAs on tile nt -> park tile nt+1 -> merge -> ONE barrier.
// stile write->merge read is wave-internal (LDS ops of a wave are in order),
// so it needs no extra barrier. Each wave's per-token top-32 list is final.
// ---------------------------------------------------------------------------
__global__ __launch_bounds__(256) void scores_topk_kernel(
    const __bf16* __restrict__ qbf, const __bf16* __restrict__ kbf,
    float* __restrict__ topk_s, int* __restrict__ topk_i) {
  __shared__ __bf16 btile[2][16 * 128];   // 2 x 4 KB staged key tiles
  __shared__ float  stile[8][16][16];     // 8 KB score tiles (per wave)
  __shared__ float  ls[8][16][TK];        // 16 KB top-32 scores
  __shared__ int    li[8][16][TK];        // 16 KB top-32 indices

  const int lane = threadIdx.x & 31;
  const int wave = threadIdx.x >> 5;
  const int tok0 = blockIdx.x * 128 + wave * 16;
  const int trow = threadIdx.x >> 4;      // 0..15 : key row in tile
  const int tch  = threadIdx.x & 15;      // 0..15 : 16B chunk in row (256B)

  for (int t = lane; t < 16 * TK; t += 32) {
    (&ls[wave][0][0])[t] = -1e30f;
    (&li[wave][0][0])[t] = 0;
  }

  v16bf afrag[4];
#pragma unroll
  for (int kk = 0; kk < 4; ++kk)
    afrag[kk] = load_frag_bf16(qbf, tok0, kk * 32, DK, lane);

  float mv = -1e30f;  // running min of this lane's token list (lanes 0..15)
  int   am = 0;

  // this thread's 16B chunk within a tile; tile stride = 256 uint4
  const uint4* ksrc = (const uint4*)kbf + (size_t)trow * 16 + tch;
  uint4 stage = ksrc[0];
  ((uint4*)btile[0])[threadIdx.x] = stage;
  __syncthreads();

  const int NT = MN / 16;  // 2048 key tiles
  for (int nt = 0; nt < NT; ++nt) {
    const int  cur  = nt & 1;
    const bool more = (nt + 1) < NT;
    if (more) {                                   // fetch next tile (in flight)
      stage = ksrc[(size_t)(nt + 1) * 256];
      if (nt + 2 < NT)
        __builtin_prefetch((const char*)(ksrc + (size_t)(nt + 2) * 256), 0, 1);
    }
    const v16bf b0 = load_frag_bf16(btile[cur], 0,  0, DK, lane);
    const v16bf b1 = load_frag_bf16(btile[cur], 0, 32, DK, lane);
    const v16bf b2 = load_frag_bf16(btile[cur], 0, 64, DK, lane);
    const v16bf b3 = load_frag_bf16(btile[cur], 0, 96, DK, lane);
    if (more) ((uint4*)btile[cur ^ 1])[threadIdx.x] = stage;

    v8f acc = {};
    acc = __builtin_amdgcn_wmma_f32_16x16x32_bf16(false, afrag[0], false, b0,
                                                  (short)0, acc, false, false);
    acc = __builtin_amdgcn_wmma_f32_16x16x32_bf16(false, afrag[1], false, b1,
                                                  (short)0, acc, false, false);
    acc = __builtin_amdgcn_wmma_f32_16x16x32_bf16(false, afrag[2], false, b2,
                                                  (short)0, acc, false, false);
    acc = __builtin_amdgcn_wmma_f32_16x16x32_bf16(false, afrag[3], false, b3,
                                                  (short)0, acc, false, false);

    const int col = lane & 15;
    const int rb  = (lane >> 4) << 3;
#pragma unroll
    for (int v = 0; v < 8; ++v) stile[wave][rb + v][col] = acc[v];

    if (lane < 16) {
      float* Ls = ls[wave][lane];
      int*   Li = li[wave][lane];
#pragma unroll 4
      for (int c = 0; c < 16; ++c) {
        const float s = stile[wave][lane][c];
        if (s > mv) {
          Ls[am] = s;
          Li[am] = nt * 16 + c;
          mv = Ls[0]; am = 0;
          for (int j = 1; j < TK; ++j)
            if (Ls[j] < mv) { mv = Ls[j]; am = j; }
        }
      }
    }
    __syncthreads();  // orders parked tile for all waves; protects btile[cur]
  }

  if (lane < 16) {
    const size_t o = (size_t)(tok0 + lane) * TK;
    for (int j = 0; j < TK; ++j) {
      topk_s[o + j] = ls[wave][lane][j];
      topk_i[o + j] = li[wave][lane][j];
    }
  }
}

// ---------------------------------------------------------------------------
// 2 hop rounds + softmax + weighted value gather. One block (256 thr)/token.
// Writes r (fp32) and bf16 r into second half of hr_bf.
// ---------------------------------------------------------------------------
__global__ __launch_bounds__(256) void hops_read_kernel(
    const float* __restrict__ qn, const __bf16* __restrict__ kbf,
    const int* __restrict__ edges, const float* __restrict__ ew,
    const float* __restrict__ nvals,
    const float* __restrict__ topk_s, const int* __restrict__ topk_i,
    float* __restrict__ r_out, __bf16* __restrict__ hr) {
  const int tok = blockIdx.x, tid = threadIdx.x;
  __shared__ float qh[DK];
  __shared__ float cs[TK];
  __shared__ int   ci[TK];
  __shared__ float cand_s[TK * EM];
  __shared__ int   cand_i[TK * EM];
  __shared__ float alpha[TK];

  if (tid < DK) qh[tid] = qn[(size_t)tok * DK + tid];
  if (tid < TK) {
    cs[tid] = topk_s[(size_t)tok * TK + tid];
    ci[tid] = topk_i[(size_t)tok * TK + tid];
  }
  __syncthreads();

  for (int hop = 0; hop < 2; ++hop) {
    const int k = tid >> 3, j = tid & (EM - 1);
    const int src = ci[k];
    const int nb  = edges[(size_t)src * EM + j];
    const float w = ew[(size_t)src * EM + j];
    const int ok  = (nb >= 0) && (nb < MN);
    const int nbc = min(max(nb, 0), MN - 1);
    const __bf16* kp = kbf + (size_t)nbc * DK;
    float sim = 0.f;
#pragma unroll 8
    for (int d = 0; d < DK; ++d) sim += qh[d] * (float)kp[d];
    __syncthreads();
    cand_s[tid] = ok ? sim * w : 0.f;
    cand_i[tid] = nbc;
    __syncthreads();
    if (tid == 0) {
      float mv = cs[0]; int am = 0;
      for (int q = 1; q < TK; ++q) if (cs[q] < mv) { mv = cs[q]; am = q; }
      for (int e = 0; e < TK * EM; ++e) {
        const float s = cand_s[e];
        if (s > mv) {
          cs[am] = s; ci[am] = cand_i[e];
          mv = cs[0]; am = 0;
          for (int q = 1; q < TK; ++q) if (cs[q] < mv) { mv = cs[q]; am = q; }
        }
      }
    }
    __syncthreads();
  }

  if (tid == 0) {
    float mx = cs[0];
    for (int j = 1; j < TK; ++j) mx = fmaxf(mx, cs[j]);
    float sum = 0.f;
    for (int j = 0; j < TK; ++j) {
      const float e = __expf((cs[j] - mx) / SCALE);
      alpha[j] = e; sum += e;
    }
    const float inv = 1.f / sum;
    for (int j = 0; j < TK; ++j) alpha[j] *= inv;
  }
  __syncthreads();

  for (int d = tid; d < DM; d += 256) {
    float acc = 0.f;
#pragma unroll 8
    for (int k = 0; k < TK; ++k)
      acc += alpha[k] * nvals[(size_t)ci[k] * DM + d];
    r_out[(size_t)tok * DM + d] = acc;
    hr[(size_t)tok * (2 * DM) + DM + d] = (__bf16)acc;
  }
}

// ---------------------------------------------------------------------------
// Surprise gate + gated residual output. One block (256 thr)/token.
// ---------------------------------------------------------------------------
__global__ __launch_bounds__(256) void gate_out_kernel(
    const float* __restrict__ h, const float* __restrict__ v,
    const float* __restrict__ r, const float* __restrict__ fused,
    const float* __restrict__ bias, const float* __restrict__ tau,
    float* __restrict__ out) {
  const int tok = blockIdx.x, tid = threadIdx.x;
  __shared__ float rd[256], rv[256];
  __shared__ float gsh;
  float ds = 0.f, vs = 0.f;
  for (int d = tid; d < DM; d += 256) {
    const float vv = v[(size_t)tok * DM + d];
    const float dr = vv - r[(size_t)tok * DM + d];
    ds += dr * dr;
    vs += vv * vv;
  }
  rd[tid] = ds; rv[tid] = vs;
  __syncthreads();
  for (int s = 128; s > 0; s >>= 1) {
    if (tid < s) { rd[tid] += rd[tid + s]; rv[tid] += rv[tid + s]; }
    __syncthreads();
  }
  if (tid == 0) {
    const float surprise = rd[0] / (rv[0] + 1e-8f);
    gsh = 1.f / (1.f + __expf(-(surprise - tau[0]) / 0.1f));
  }
  __syncthreads();
  const float g = gsh;
  for (int d = tid; d < DM; d += 256) {
    const size_t o = (size_t)tok * DM + d;
    out[o] = h[o] + g * (fused[o] + bias[d]);
  }
}

// ---------------------------------------------------------------------------
// Host launch
// ---------------------------------------------------------------------------
extern "C" void kernel_launch(void* const* d_in, const int* in_sizes, int n_in,
                              void* d_out, int out_size, void* d_ws, size_t ws_size,
                              hipStream_t stream) {
  const float* h      = (const float*)d_in[0];
  const float* W_q    = (const float*)d_in[1];
  /* d_in[2] = W_k (unused by reference) */
  const float* W_v    = (const float*)d_in[3];
  const float* W_mu_w = (const float*)d_in[4];
  const float* W_mu_b = (const float*)d_in[5];
  const float* nkeys  = (const float*)d_in[6];
  const float* nvals  = (const float*)d_in[7];
  const float* ew     = (const float*)d_in[8];
  const float* tau    = (const float*)d_in[9];
  const int*   edges  = (const int*)d_in[10];
  float* out = (float*)d_out;

  char* ws = (char*)d_ws;
  size_t off = 0;
  auto take = [&](size_t bytes) { char* p = ws + off; off = (off + bytes + 255) & ~(size_t)255; return p; };
  __bf16* wq_bf  = (__bf16*)take((size_t)DK * DM * 2);
  __bf16* wv_bf  = (__bf16*)take((size_t)DM * DM * 2);
  __bf16* wmu_bf = (__bf16*)take((size_t)DM * 2 * DM * 2);
  __bf16* k_bf   = (__bf16*)take((size_t)MN * DK * 2);
  __bf16* hr_bf  = (__bf16*)take((size_t)NTOK * 2 * DM * 2);
  float*  q_pre  = (float*)take((size_t)NTOK * DK * 4);
  float*  q_nrm  = (float*)take((size_t)NTOK * DK * 4);
  __bf16* q_bf   = (__bf16*)take((size_t)NTOK * DK * 2);
  float*  v_f    = (float*)take((size_t)NTOK * DM * 4);
  float*  r_f    = (float*)take((size_t)NTOK * DM * 4);
  float*  fu_f   = (float*)take((size_t)NTOK * DM * 4);
  float*  tk_s   = (float*)take((size_t)NTOK * TK * 4);
  int*    tk_i   = (int*)take((size_t)NTOK * TK * 4);
  (void)ws_size; (void)in_sizes; (void)n_in; (void)out_size;

  // 1) precision conversion
  f2bf_kernel<<<256, 256, 0, stream>>>(W_q, wq_bf, DK * DM);
  f2bf_kernel<<<1024, 256, 0, stream>>>(W_v, wv_bf, DM * DM);
  f2bf_kernel<<<2048, 256, 0, stream>>>(W_mu_w, wmu_bf, DM * 2 * DM);
  h_to_hr_kernel<<<4096, 256, 0, stream>>>(h, hr_bf);

  // 2) normalize node keys
  norm_keys_kernel<<<MN, 128, 0, stream>>>(nkeys, k_bf);

  // 3) q = h W_q^T : (4096/128)*(128/64) = 64 blocks
  wmma_gemm_bf16<<<64, 256, 0, stream>>>(hr_bf, 2 * DM, wq_bf, DM,
                                         q_pre, DK, NTOK, DK, DM);
  norm_q_kernel<<<NTOK, 128, 0, stream>>>(q_pre, q_nrm, q_bf);

  // 4) v = h W_v^T : (4096/128)*(1024/64) = 512 blocks
  wmma_gemm_bf16<<<512, 256, 0, stream>>>(hr_bf, 2 * DM, wv_bf, DM,
                                          v_f, DM, NTOK, DM, DM);

  // 5) fused scores (4096x32768x128) + top-32 : 4096/128 = 32 blocks
  scores_topk_kernel<<<NTOK / 128, 256, 0, stream>>>(q_bf, k_bf, tk_s, tk_i);

  // 6) 2 hop rounds + softmax + value read -> r, hr second half
  hops_read_kernel<<<NTOK, 256, 0, stream>>>(q_nrm, k_bf, edges, ew, nvals,
                                             tk_s, tk_i, r_f, hr_bf);

  // 7) fused = [h;r] W_mu^T (K = 2048) : 512 blocks
  wmma_gemm_bf16<<<512, 256, 0, stream>>>(hr_bf, 2 * DM, wmu_bf, 2 * DM,
                                          fu_f, DM, NTOK, DM, 2 * DM);

  // 8) surprise gate + residual
  gate_out_kernel<<<NTOK, 256, 0, stream>>>(h, v_f, r_f, fu_f, W_mu_b, tau, out);
}
